// energy_point_residual_19842748908338
// MI455X (gfx1250) — compile-verified, matrix operand order checked by reading, not run on hardware
//
#include <hip/hip_runtime.h>
#include <hip/hip_bf16.h>

typedef __attribute__((ext_vector_type(16))) _Float16 v16h;
typedef __attribute__((ext_vector_type(8)))  _Float16 v8h;
typedef __attribute__((ext_vector_type(8)))  float    v8f;

#define NPTS 4096
#define NB   8
#define KNN  16
#define DSTRIDE 4100   // padded LDS row stride: shifts rows m vs m+8 by 32 banks

// ---------------------------------------------------------------------------
// WMMA 16x16x32 f16 fragment loader (wave32).
//   lane L in [0,15]  -> row = rbase + L, halves 0..7  = k[koff..koff+7],
//                                        halves 8..15 = k[koff+16..koff+23]
//   lane L in [16,31] -> row = rbase + L-16, same but koff += 8.
// Both halves are 16B-contiguous -> two global_load_b128 per lane.
// ---------------------------------------------------------------------------
__device__ __forceinline__ v16h load_frag(const _Float16* __restrict__ base,
                                          int row, int stride, int koff) {
  const _Float16* p = base + (size_t)row * stride + koff;
  v8h lo = *reinterpret_cast<const v8h*>(p);
  v8h hi = *reinterpret_cast<const v8h*>(p + 16);
  v16h f;
  f[0]=lo[0]; f[1]=lo[1]; f[2]=lo[2]; f[3]=lo[3];
  f[4]=lo[4]; f[5]=lo[5]; f[6]=lo[6]; f[7]=lo[7];
  f[8]=hi[0]; f[9]=hi[1]; f[10]=hi[2]; f[11]=hi[3];
  f[12]=hi[4]; f[13]=hi[5]; f[14]=hi[6]; f[15]=hi[7];
  return f;
}

__device__ __forceinline__ v8f wmma_f16(v16h a, v16h b, v8f c) {
  return __builtin_amdgcn_wmma_f32_16x16x32_f16(false, a, false, b,
                                                (short)0, c, false, false);
}

// ---------------------------------------------------------------------------
__global__ void k_cvt_f16(const float* __restrict__ in, _Float16* __restrict__ out, int n) {
  int i = blockIdx.x * 256 + threadIdx.x;
  if (i < n) out[i] = (_Float16)in[i];
}

// ---------------------------------------------------------------------------
// conv0: [8,3,4096] x w0[64,3] + b0 -> yraw [8][64][4096] f32 (channel-major)
// ---------------------------------------------------------------------------
__global__ void k_conv0(const float* __restrict__ x, const float* __restrict__ w,
                        const float* __restrict__ bias, float* __restrict__ y) {
  int i = blockIdx.x * 256 + threadIdx.x;     // b*4096 + n
  int b = i >> 12, n = i & (NPTS - 1);
  const float* xb = x + (size_t)b * 3 * NPTS + n;
  float i0 = xb[0], i1 = xb[NPTS], i2 = xb[2 * NPTS];
  float* yb = y + (size_t)b * 64 * NPTS + n;
#pragma unroll
  for (int c = 0; c < 64; c++) {
    yb[(size_t)c * NPTS] = fmaf(w[c*3], i0, fmaf(w[c*3+1], i1, fmaf(w[c*3+2], i2, bias[c])));
  }
}

// ---------------------------------------------------------------------------
// LayerNorm over the point axis (per (b,channel) row) + ReLU, transposing to
// point-major f16 [b][n][C]. grid = B*C blocks, 256 threads.
// ---------------------------------------------------------------------------
__global__ void k_ln_relu_t(const float* __restrict__ y, const float* __restrict__ g,
                            const float* __restrict__ be, _Float16* __restrict__ xout,
                            int C) {
  int row = blockIdx.x;                 // b*C + c
  int c   = row % C;
  size_t b = (size_t)(row / C);
  const float* p = y + (size_t)row * NPTS;
  int t = threadIdx.x;

  float vals[16];
  float s = 0.f, s2 = 0.f;
#pragma unroll
  for (int i = 0; i < 16; i++) {
    float v = p[t + i * 256];
    vals[i] = v; s += v; s2 += v * v;
  }
#pragma unroll
  for (int off = 16; off; off >>= 1) {
    s  += __shfl_xor(s,  off, 32);
    s2 += __shfl_xor(s2, off, 32);
  }
  __shared__ float ws1[8], ws2[8];
  int wid = t >> 5, lane = t & 31;
  if (lane == 0) { ws1[wid] = s; ws2[wid] = s2; }
  __syncthreads();
  if (t == 0) {
    float a = 0.f, bb = 0.f;
#pragma unroll
    for (int i = 0; i < 8; i++) { a += ws1[i]; bb += ws2[i]; }
    ws1[0] = a; ws2[0] = bb;
  }
  __syncthreads();
  float mu   = ws1[0] * (1.f / NPTS);
  float var  = ws2[0] * (1.f / NPTS) - mu * mu;
  float rstd = rsqrtf(var + 1e-5f);
#pragma unroll
  for (int i = 0; i < 16; i++) {
    int n = t + i * 256;
    float v = fmaf((vals[i] - mu) * rstd, g[n], be[n]);
    v = fmaxf(v, 0.f);
    xout[(b * NPTS + n) * (size_t)C + c] = (_Float16)v;
  }
}

// ---------------------------------------------------------------------------
// WMMA GEMM: Y[b][co][n] = sum_k Wh[co][k] * X[b][n][k] + bias[co]
// grid = (N/128, Cout/16, B); block 256 (8 waves); one 16x16 tile per wave.
// CIN is compile-time so the K-loop fully unrolls.
// ---------------------------------------------------------------------------
template<int CIN>
__global__ void k_gemm_wmma(const _Float16* __restrict__ Wh, const float* __restrict__ bias,
                            const _Float16* __restrict__ X, float* __restrict__ Y,
                            int Cout) {
  int wid  = threadIdx.x >> 5, lane = threadIdx.x & 31;
  int b    = blockIdx.z;
  int cob  = blockIdx.y * 16;
  int pb   = (blockIdx.x * 8 + wid) * 16;
  int rsel = lane & 15;
  int ksel = (lane >> 4) << 3;

  const _Float16* Xb = X + (size_t)b * NPTS * CIN;
  v8f acc = {};
#pragma unroll
  for (int kc = 0; kc < CIN; kc += 32) {
    v16h a  = load_frag(Wh, cob + rsel, CIN, kc + ksel);
    v16h bf = load_frag(Xb, pb  + rsel, CIN, kc + ksel);
    acc = wmma_f16(a, bf, acc);
  }
  int mofs = (lane >> 4) << 3;
  float* Yb = Y + (size_t)b * Cout * NPTS + (pb + rsel);
#pragma unroll
  for (int v = 0; v < 8; v++) {
    int co = cob + mofs + v;
    Yb[(size_t)co * NPTS] = acc[v] + bias[co];
  }
}

// ---------------------------------------------------------------------------
__global__ void k_sqnorm(const _Float16* __restrict__ f, float* __restrict__ sq) {
  int p = blockIdx.x * 256 + threadIdx.x;      // b*4096+n
  const _Float16* fp = f + (size_t)p * 256;
  float s = 0.f;
#pragma unroll 8
  for (int c = 0; c < 256; c++) { float v = (float)fp[c]; s = fmaf(v, v, s); }
  sq[p] = s;
}

// ---------------------------------------------------------------------------
// Edge-conv: per (batch, 16-row tile):
//  Phase 1: WMMA Gram strip 16x4096 -> distances in 262KB padded LDS strip
//  Phase 2: lazy-rescan top-16 per row (only argmin-winning lane rescans)
//  Phase 3: gather 16 neighbors, channel max, atomic-accumulate mean pool
// ---------------------------------------------------------------------------
__global__ void k_edgeconv(const _Float16* __restrict__ feat, const float* __restrict__ sq,
                           float* __restrict__ pooled) {
  extern __shared__ float sdist[];             // [16][DSTRIDE] floats
  __shared__ int   s_knn[16][KNN];
  __shared__ float s_sqn[16];

  int wid  = threadIdx.x >> 5, lane = threadIdx.x & 31;
  int b    = blockIdx.y;
  int rowb = blockIdx.x * 16;
  int rsel = lane & 15;
  int ksel = (lane >> 4) << 3;
  const _Float16* fb = feat + (size_t)b * NPTS * 256;
  const float*    sb = sq + (size_t)b * NPTS;

  if (threadIdx.x < 16) s_sqn[threadIdx.x] = sb[rowb + threadIdx.x];
  __syncthreads();

  // -- Phase 1: Gram strip via WMMA (K=256 -> 8 chunks) --------------------
  v16h a[8];
#pragma unroll
  for (int k = 0; k < 8; k++)
    a[k] = load_frag(fb, rowb + rsel, 256, k * 32 + ksel);

  int mofs = (lane >> 4) << 3;
  for (int ct = wid; ct < NPTS / 16; ct += 8) {
    int colb = ct * 16;
    if (ct + 8 < NPTS / 16)
      __builtin_prefetch(fb + (size_t)(colb + 128) * 256, 0, 1);
    v8f acc = {};
#pragma unroll
    for (int k = 0; k < 8; k++) {
      v16h bf = load_frag(fb, colb + rsel, 256, k * 32 + ksel);
      acc = wmma_f16(a[k], bf, acc);
    }
    float sqm = sb[colb + rsel];
#pragma unroll
    for (int v = 0; v < 8; v++) {
      int m = mofs + v;
      sdist[m * DSTRIDE + colb + rsel] = s_sqn[m] + sqm - 2.f * acc[v];
    }
  }
  __syncthreads();

  // -- Phase 2: lazy-rescan top-16 per row ---------------------------------
  // Each lane owns a disjoint strided slice; after extracting the global
  // argmin, only the winning lane's cached (min,idx) is stale.
  for (int r = wid; r < 16; r += 8) {
    float* drow = sdist + (size_t)r * DSTRIDE;
    float bv = 3.4e38f; int bi = -1;
    for (int c = lane; c < NPTS; c += 32) {
      float v = drow[c];
      if (v < bv) { bv = v; bi = c; }
    }
    for (int it = 0; it < KNN; it++) {
      float gv = bv; int gi = bi;
#pragma unroll
      for (int off = 16; off; off >>= 1) {
        float ov = __shfl_xor(gv, off, 32);
        int   oi = __shfl_xor(gi, off, 32);
        if (ov < gv || (ov == gv && oi < gi)) { gv = ov; gi = oi; }
      }
      if (lane == 0) s_knn[r][it] = gi;
      drow[gi] = 3.4e38f;                 // uniform store; in-wave DS ordering
      if (it + 1 < KNN && bi == gi) {     // only the winning lane rescans
        bv = 3.4e38f; bi = -1;
        for (int c = lane; c < NPTS; c += 32) {
          float v = drow[c];
          if (v < bv) { bv = v; bi = c; }
        }
      }
    }
  }
  __syncthreads();

  // -- Phase 3: gather + max over K, accumulate mean pool ------------------
  int c = threadIdx.x;                          // channel 0..255
  float accum = 0.f;
  for (int r = 0; r < 16; r++) {
    float m = -3.4e38f;
#pragma unroll
    for (int k = 0; k < KNN; k++) {
      int idx = s_knn[r][k];
      m = fmaxf(m, (float)fb[(size_t)idx * 256 + c]);
    }
    accum += m;
  }
  atomicAdd(&pooled[b * 256 + c], accum * (1.f / NPTS));
}

// ---------------------------------------------------------------------------
// Final MLP: 256 -> 128 -> 64 -> 1 per batch, single block.
// ---------------------------------------------------------------------------
__global__ void k_mlp(const float* __restrict__ pooled,
                      const float* __restrict__ gw0, const float* __restrict__ gb0,
                      const float* __restrict__ gw1, const float* __restrict__ gb1,
                      const float* __restrict__ gw2, const float* __restrict__ gb2,
                      float* __restrict__ out) {
  __shared__ float sp[NB * 256], sh1[NB * 128], sh2[NB * 64];
  int t = threadIdx.x;
  for (int i = t; i < NB * 256; i += 256) sp[i] = pooled[i];
  __syncthreads();
  for (int o = t; o < NB * 128; o += 256) {
    int b = o >> 7, j = o & 127;
    float acc = gb0[j];
    for (int k = 0; k < 256; k++) acc = fmaf(gw0[j * 256 + k], sp[b * 256 + k], acc);
    sh1[o] = fmaxf(acc, 0.f);
  }
  __syncthreads();
  for (int o = t; o < NB * 64; o += 256) {
    int b = o >> 6, j = o & 63;
    float acc = gb1[j];
    for (int k = 0; k < 128; k++) acc = fmaf(gw1[j * 128 + k], sh1[b * 128 + k], acc);
    sh2[o] = fmaxf(acc, 0.f);
  }
  __syncthreads();
  if (t < NB) {
    float acc = gb2[0];
    for (int k = 0; k < 64; k++) acc = fmaf(gw2[k], sh2[t * 64 + k], acc);
    out[t] = acc;
  }
}

// ---------------------------------------------------------------------------
extern "C" void kernel_launch(void* const* d_in, const int* in_sizes, int n_in,
                              void* d_out, int out_size, void* d_ws, size_t ws_size,
                              hipStream_t stream) {
  const float* pc  = (const float*)d_in[0];
  const float* w0  = (const float*)d_in[1];
  const float* b0  = (const float*)d_in[2];
  const float* g0  = (const float*)d_in[3];
  const float* be0 = (const float*)d_in[4];
  const float* w1  = (const float*)d_in[5];
  const float* b1  = (const float*)d_in[6];
  const float* g1  = (const float*)d_in[7];
  const float* be1 = (const float*)d_in[8];
  const float* w2  = (const float*)d_in[9];
  const float* b2  = (const float*)d_in[10];
  const float* g2  = (const float*)d_in[11];
  const float* be2 = (const float*)d_in[12];
  const float* gw0 = (const float*)d_in[13];
  const float* gb0 = (const float*)d_in[14];
  const float* gw1 = (const float*)d_in[15];
  const float* gb1 = (const float*)d_in[16];
  const float* gw2 = (const float*)d_in[17];
  const float* gb2 = (const float*)d_in[18];
  float* out = (float*)d_out;

  // Workspace carve-up (all offsets MiB/256B aligned); total ~60.2 MiB.
  char* ws = (char*)d_ws;
  size_t off = 0;
  _Float16* xA  = (_Float16*)(ws + off); off += (size_t)NB * NPTS *  64 * 2;  // 4 MiB
  _Float16* xB  = (_Float16*)(ws + off); off += (size_t)NB * NPTS * 128 * 2;  // 8 MiB
  _Float16* xC  = (_Float16*)(ws + off); off += (size_t)NB * NPTS * 256 * 2;  // 16 MiB
  float*    yraw= (float*)   (ws + off); off += (size_t)NB * 256 * NPTS * 4;  // 32 MiB
  _Float16* w1h = (_Float16*)(ws + off); off += 128 * 64 * 2;
  _Float16* w2h = (_Float16*)(ws + off); off += 256 * 128 * 2;
  float*    sqn = (float*)   (ws + off); off += (size_t)NB * NPTS * 4;
  float*    pooled = (float*)(ws + off); off += NB * 256 * 4;

  // weights -> f16
  k_cvt_f16<<<(128 * 64  + 255) / 256, 256, 0, stream>>>(w1, w1h, 128 * 64);
  k_cvt_f16<<<(256 * 128 + 255) / 256, 256, 0, stream>>>(w2, w2h, 256 * 128);

  // conv0 + LN/ReLU (-> point-major f16)
  k_conv0<<<NB * NPTS / 256, 256, 0, stream>>>(pc, w0, b0, yraw);
  k_ln_relu_t<<<NB * 64, 256, 0, stream>>>(yraw, g0, be0, xA, 64);

  // conv1 (WMMA) + LN/ReLU
  k_gemm_wmma<64><<<dim3(NPTS / 128, 128 / 16, NB), 256, 0, stream>>>(w1h, b1, xA, yraw, 128);
  k_ln_relu_t<<<NB * 128, 256, 0, stream>>>(yraw, g1, be1, xB, 128);

  // conv2 (WMMA) + LN/ReLU
  k_gemm_wmma<128><<<dim3(NPTS / 128, 256 / 16, NB), 256, 0, stream>>>(w2h, b2, xB, yraw, 256);
  k_ln_relu_t<<<NB * 256, 256, 0, stream>>>(yraw, g2, be2, xC, 256);

  // squared norms, zero the pooled accumulator, edge-conv + mean pool
  k_sqnorm<<<NB * NPTS / 256, 256, 0, stream>>>(xC, sqn);
  hipMemsetAsync(pooled, 0, NB * 256 * sizeof(float), stream);
  k_edgeconv<<<dim3(NPTS / 16, NB), 256, 16 * DSTRIDE * sizeof(float), stream>>>(xC, sqn, pooled);

  // final MLP -> [8,1]
  k_mlp<<<1, 256, 0, stream>>>(pooled, gw0, gb0, gw1, gb1, gw2, gb2, out);
  (void)in_sizes; (void)n_in; (void)out_size; (void)ws_size;
}
